// RotatedVariedSizeWindowAttention_10849087390005
// MI455X (gfx1250) — matrix operand order repaired and checked by reading, not compiled, not run on hardware
//
#include <hip/hip_runtime.h>
#include <hip/hip_bf16.h>

typedef __attribute__((ext_vector_type(16))) _Float16 v16h;
typedef __attribute__((ext_vector_type(8)))  float    v8f;

#define NH 12
#define HD 64
#define CCH 768
#define NPIX 4096
#define NEG_SLOPE 0.01f

// ---------------- WMMA helpers (wave32, CDNA5) ----------------
__device__ inline v8f wmma_f32_f16(v16h a, v16h b, v8f c) {
  // v_wmma_f32_16x16x32_f16: 8 args (neg_a, A, neg_b, B, c_mod, C, reuse_a, reuse_b)
  return __builtin_amdgcn_wmma_f32_16x16x32_f16(false, a, false, b, (short)0, c, false, false);
}

// ---------------- CDNA5 async global->LDS (ASYNCcnt path) ----------------
// GV mode: per-lane 64-bit global address, per-lane 32-bit LDS byte address.
__device__ inline unsigned lds_addr_of(const void* p) {
  // generic shared address: aperture in [63:32], LDS byte offset in [31:0]
  return (unsigned)(size_t)p;
}
__device__ inline void async_g2l_b128(unsigned lds_addr, const void* g) {
  asm volatile("global_load_async_to_lds_b128 %0, %1, off"
               :: "v"(lds_addr), "v"(g) : "memory");
}
__device__ inline void wait_async0() {
  asm volatile("s_wait_asynccnt 0x0" ::: "memory");
}

// A fragment 16x32 f16 from LDS, layout [row][K] with row stride `stride` halves.
// Lane l: row = row0 + (l&15); first 8 halves K = hi*8+0..7, next 8: K = 16+hi*8+0..7.
__device__ inline v16h lds_load_afrag(const _Float16* lds, int stride, int row0, int k0) {
  int lane = threadIdx.x & 31;
  int row = row0 + (lane & 15);
  int hi  = lane >> 4;
  const _Float16* p = lds + row * stride + k0 + hi * 8;
  v16h a;
#pragma unroll
  for (int i = 0; i < 8; ++i) { a[i] = p[i]; a[8 + i] = p[16 + i]; }
  return a;
}

// B fragment 32x16 f16 from LDS, layout [col][K] with col stride `stride` halves.
// Lane l: col = col0 + (l&15); first 8 halves K = hi*8+0..7, next 8: K = 16+hi*8+0..7.
__device__ inline v16h lds_load_bfrag(const _Float16* lds, int stride, int col0, int k0) {
  int lane = threadIdx.x & 31;
  int col = col0 + (lane & 15);
  int kb  = (lane >> 4) * 8;
  const _Float16* p = lds + col * stride + k0 + kb;
  v16h b;
#pragma unroll
  for (int i = 0; i < 8; ++i) { b[i] = p[i]; b[8 + i] = p[16 + i]; }
  return b;
}

// ---------------- 1: window pooling + leaky relu ----------------
__global__ void pool_kernel(const float* __restrict__ x, float* __restrict__ pooled) {
  int blk = blockIdx.x;             // b*64 + widx
  int b = blk >> 6, widx = blk & 63;
  int wy = widx >> 3, wx = widx & 7;
  for (int c = threadIdx.x; c < CCH; c += blockDim.x) {
    float s = 0.f;
    const float* xb = x + (size_t)b * NPIX * CCH + c;
#pragma unroll 4
    for (int i = 0; i < 8; ++i)
      for (int j = 0; j < 8; ++j) {
        int pix = ((wy * 8 + i) << 6) + wx * 8 + j;
        s += xb[(size_t)pix * CCH];
      }
    s *= (1.f / 64.f);
    s = s < 0.f ? s * NEG_SLOPE : s;
    pooled[(size_t)blk * CCH + c] = s;
  }
}

// ---------------- 2: conv1x1 -> offset/scale/angle params ----------------
// prm layout: [p=b*12+h][slot: off0,off1,sc0,sc1,ang][widx]
__global__ void conv_kernel(const float* __restrict__ pooled,
                            const float* __restrict__ wo, const float* __restrict__ bo,
                            const float* __restrict__ wsc, const float* __restrict__ bsc,
                            const float* __restrict__ wa, const float* __restrict__ ba,
                            float* __restrict__ prm) {
  __shared__ float red[60 * 64];
  int blk = blockIdx.x;             // b*64 + widx
  int b = blk >> 6, widx = blk & 63;
  int tid = threadIdx.x;            // 64 threads
  float acc[60];
#pragma unroll
  for (int o = 0; o < 60; ++o) acc[o] = 0.f;
  const float* pv = pooled + (size_t)blk * CCH;
  for (int c = tid; c < CCH; c += 64) {
    float p = pv[c];
#pragma unroll
    for (int o = 0; o < 24; ++o) acc[o]      += p * wo[o * CCH + c];
#pragma unroll
    for (int o = 0; o < 24; ++o) acc[24 + o] += p * wsc[o * CCH + c];
#pragma unroll
    for (int o = 0; o < 12; ++o) acc[48 + o] += p * wa[o * CCH + c];
  }
#pragma unroll
  for (int o = 0; o < 60; ++o) red[o * 64 + tid] = acc[o];
  __syncthreads();
  if (tid < 60) {
    float s = 0.f;
    for (int t = 0; t < 64; ++t) s += red[tid * 64 + t];
    int o = tid, p, slot; float bias;
    if (o < 24)      { p = b * NH + (o >> 1); slot = o & 1; bias = bo[o]; }
    else if (o < 48) { int oo = o - 24; p = b * NH + (oo >> 1); slot = 2 + (oo & 1); bias = bsc[oo]; }
    else             { int oo = o - 48; p = b * NH + oo; slot = 4; bias = ba[oo]; }
    prm[((size_t)p * 5 + slot) * 64 + widx] = s + bias;
  }
}

// ---------------- 3: rotated/scaled sample coordinates ----------------
__global__ void coords_kernel(const float* __restrict__ prm, float* __restrict__ coords, int total) {
  int gid = blockIdx.x * blockDim.x + threadIdx.x;  // p*4096 + lh*64 + lw
  if (gid >= total) return;
  int p = gid >> 12; int rem = gid & 4095;
  int lh = rem >> 6, lw = rem & 63;
  int wy = lh >> 3, ai = lh & 7, wx = lw >> 3, aj = lw & 7;
  int widx = wy * 8 + wx;
  const float* pp = prm + (size_t)p * 5 * 64;
  float off0 = pp[0 * 64 + widx] * 0.125f;   // / (H//ws)
  float off1 = pp[1 * 64 + widx] * 0.125f;   // / (W//ws)
  float sc0  = pp[2 * 64 + widx];
  float sc1  = pp[3 * 64 + widx];
  float ang  = pp[4 * 64 + widx];
  float wrx = -1.f + 2.f * (wx * 8 + 3.5f) / 63.f;   // window-center refs
  float wry = -1.f + 2.f * (wy * 8 + 3.5f) / 63.f;
  float bcw = (aj - 3.5f) * (2.f / 63.f);            // centered in-window grid
  float bch = (ai - 3.5f) * (2.f / 63.f);
  float wcs0 = bcw * (sc0 + 1.f);
  float wcs1 = bch * (sc1 + 1.f);
  float sn = __sinf(ang), cs = __cosf(ang);
  float cx = wrx + (-wcs1 * sn + wcs0 * cs) + off0;
  float cy = wry + ( wcs1 * cs + wcs0 * sn) + off1;
  coords[(size_t)gid * 2 + 0] = cx;
  coords[(size_t)gid * 2 + 1] = cy;
}

// ---------------- 4: QKV GEMM (M=B*4096, N=2304, K=768) ----------------
// f32 inputs must pass through VGPRs for f16 conversion -> synchronous staging.
#define TK 32
#define LSTR 40
__global__ __launch_bounds__(256) void qkv_gemm_kernel(
    const float* __restrict__ x, const float* __restrict__ w, const float* __restrict__ bias,
    _Float16* __restrict__ q_win, _Float16* __restrict__ k_img, _Float16* __restrict__ v_img) {
  __shared__ _Float16 As[128 * LSTR];
  __shared__ _Float16 Bs[128 * LSTR];
  int bm = blockIdx.x, bn = blockIdx.y;
  int tid = threadIdx.x;
  int wid = tid >> 5;
  int waveM = (wid >> 1) * 32, waveN = (wid & 1) * 64;
  v8f zero = {};
  v8f acc[2][4];
#pragma unroll
  for (int i = 0; i < 2; ++i)
#pragma unroll
    for (int j = 0; j < 4; ++j) acc[i][j] = zero;

  int lrow = tid >> 1, lseg = (tid & 1) * 16;
  const float* ag = x + (size_t)(bm * 128 + lrow) * CCH + lseg;
  const float* bg = w + (size_t)(bn * 128 + lrow) * CCH + lseg;
  _Float16* asl = As + lrow * LSTR + lseg;
  _Float16* bsl = Bs + lrow * LSTR + lseg;

  for (int k0 = 0; k0 < CCH; k0 += TK) {
    __syncthreads();
#pragma unroll
    for (int i = 0; i < 16; ++i) asl[i] = (_Float16)ag[k0 + i];
#pragma unroll
    for (int i = 0; i < 16; ++i) bsl[i] = (_Float16)bg[k0 + i];
    __syncthreads();
    v16h af[2], bf[4];
#pragma unroll
    for (int i = 0; i < 2; ++i) af[i] = lds_load_afrag(As, LSTR, waveM + i * 16, 0);
#pragma unroll
    for (int j = 0; j < 4; ++j) bf[j] = lds_load_bfrag(Bs, LSTR, waveN + j * 16, 0);
#pragma unroll
    for (int i = 0; i < 2; ++i)
#pragma unroll
      for (int j = 0; j < 4; ++j) acc[i][j] = wmma_f32_f16(af[i], bf[j], acc[i][j]);
  }
  // epilogue: bias, fold softmax scale into Q, scatter to layouts
  int lane = tid & 31, hi = lane >> 4, lc = lane & 15;
#pragma unroll
  for (int i = 0; i < 2; ++i)
#pragma unroll
    for (int j = 0; j < 4; ++j)
#pragma unroll
      for (int r = 0; r < 8; ++r) {
        int m = bm * 128 + waveM + i * 16 + r + hi * 8;
        int n = bn * 128 + waveN + j * 16 + lc;
        float val = acc[i][j][r] + bias[n];
        int b = m >> 12, pix = m & 4095;
        if (n < 768) {                       // Q -> window layout, pre-scaled
          int head = n >> 6, d = n & 63;
          int y = pix >> 6, xw = pix & 63;
          int win = b * 64 + (y >> 3) * 8 + (xw >> 3);
          int t = (y & 7) * 8 + (xw & 7);
          q_win[(((size_t)win * NH + head) << 12) + t * 64 + d] = (_Float16)(val * 0.125f);
        } else if (n < 1536) {               // K -> image layout
          int nn = n - 768; int head = nn >> 6, d = nn & 63;
          k_img[(((size_t)(b * NH + head) * 64 + d) << 12) + pix] = (_Float16)val;
        } else {                             // V -> image layout
          int nn = n - 1536; int head = nn >> 6, d = nn & 63;
          v_img[(((size_t)(b * NH + head) * 64 + d) << 12) + pix] = (_Float16)val;
        }
      }
}

// ---------------- 5: bilinear grid sample of K and V ----------------
__global__ __launch_bounds__(256) void gridsample_kernel(
    const float* __restrict__ coords,
    const _Float16* __restrict__ k_img, const _Float16* __restrict__ v_img,
    _Float16* __restrict__ ksel, _Float16* __restrict__ vsel) {
  int tid = threadIdx.x;
  int lw = tid & 63, dq = tid >> 6;
  int blk = blockIdx.x;                  // p*1024 + lh*16 + dblk
  int p = blk >> 10; int rem = blk & 1023;
  int lh = rem >> 4; int d = ((rem & 15) << 2) + dq;
  const float* cp = coords + (((size_t)p << 12) + (lh << 6) + lw) * 2;
  float gx = (cp[0] + 1.f) * 0.5f * 63.f;   // align_corners=True
  float gy = (cp[1] + 1.f) * 0.5f * 63.f;
  float x0f = floorf(gx), y0f = floorf(gy);
  int x0 = (int)x0f, y0 = (int)y0f;
  float fx = gx - x0f, fy = gy - y0f;
  float wgt[4] = { (1.f - fx) * (1.f - fy), fx * (1.f - fy), (1.f - fx) * fy, fx * fy };
  const _Float16* kb = k_img + (((size_t)p * 64 + d) << 12);
  const _Float16* vb = v_img + (((size_t)p * 64 + d) << 12);
  float kacc = 0.f, vacc = 0.f;
#pragma unroll
  for (int c = 0; c < 4; ++c) {
    int xi = x0 + (c & 1), yi = y0 + (c >> 1);
    if (xi >= 0 && xi < 64 && yi >= 0 && yi < 64) {   // zeros padding
      int o = (yi << 6) + xi;
      kacc += wgt[c] * (float)kb[o];
      vacc += wgt[c] * (float)vb[o];
    }
  }
  int wy = lh >> 3, ai = lh & 7, wx = lw >> 3, aj = lw & 7;
  int bidx = p / NH, h = p % NH;
  int win = bidx * 64 + wy * 8 + wx;
  int kt = ai * 8 + aj;
  size_t base = ((size_t)win * NH + h) << 12;
  ksel[base + kt * 64 + d] = (_Float16)kacc;   // [kt][d] (B-frag of QK^T)
  vsel[base + d * 64 + kt] = (_Float16)vacc;   // [d][kt] (B-frag of P*V)
}

// ---------------- 6: per-(window,head) attention ----------------
#define QSTR 72
__global__ __launch_bounds__(128) void attn_kernel(
    const _Float16* __restrict__ q_win, const _Float16* __restrict__ ksel,
    const _Float16* __restrict__ vsel, const float* __restrict__ rel_table,
    _Float16* __restrict__ attn_buf) {
  __shared__ _Float16 qs[64 * QSTR];
  __shared__ _Float16 ks[64 * QSTR];
  __shared__ _Float16 vs[64 * QSTR];
  __shared__ _Float16 ps[64 * QSTR];
  int wh = blockIdx.x;                 // win*12 + head
  int head = wh % NH, win = wh / NH;
  int tid = threadIdx.x;
  size_t base = (size_t)wh << 12;
  // Async global->LDS staging of the 3 fp16 tiles (8KB each) on the ASYNCcnt path.
  // chunk c covers halves [c*8, c*8+8): row r=c>>3, col=(c&7)*8; LDS row stride 144B.
  unsigned q_l = lds_addr_of(qs), k_l = lds_addr_of(ks), v_l = lds_addr_of(vs);
#pragma unroll
  for (int it = 0; it < 4; ++it) {
    int c = tid + it * 128;            // 0..511
    int r = c >> 3, col = (c & 7) * 8;
    unsigned loff = (unsigned)((r * QSTR + col) * 2);
    size_t goff = (size_t)c * 8;
    async_g2l_b128(q_l + loff, q_win + base + goff);
    async_g2l_b128(k_l + loff, ksel + base + goff);
    async_g2l_b128(v_l + loff, vsel + base + goff);
  }
  wait_async0();
  __syncthreads();
  int wv = tid >> 5, m0 = wv * 16;
  int lane = tid & 31, hi = lane >> 4, lc = lane & 15;
  v8f acc[4];
  // init accumulators with relative-position bias: idx=(qi-ki+7)*15+(qj-kj+7)
#pragma unroll
  for (int j = 0; j < 4; ++j)
#pragma unroll
    for (int r = 0; r < 8; ++r) {
      int t = m0 + r + hi * 8, kt = j * 16 + lc;
      int ridx = ((t >> 3) - (kt >> 3) + 7) * 15 + ((t & 7) - (kt & 7) + 7);
      acc[j][r] = rel_table[ridx * NH + head];
    }
  // S = (Q*scale) @ K^T + bias
#pragma unroll
  for (int kk = 0; kk < 64; kk += 32) {
    v16h af = lds_load_afrag(qs, QSTR, m0, kk);
#pragma unroll
    for (int j = 0; j < 4; ++j) {
      v16h bf = lds_load_bfrag(ks, QSTR, j * 16, kk);
      acc[j] = wmma_f32_f16(af, bf, acc[j]);
    }
  }
  // row softmax: a row's 64 cols live on 16 lanes x 4 tiles -> xor-shuffle reduce
  float mx[8], sm[8];
#pragma unroll
  for (int r = 0; r < 8; ++r) {
    float m = fmaxf(fmaxf(acc[0][r], acc[1][r]), fmaxf(acc[2][r], acc[3][r]));
#pragma unroll
    for (int msk = 1; msk < 16; msk <<= 1) m = fmaxf(m, __shfl_xor(m, msk, 32));
    mx[r] = m;
  }
#pragma unroll
  for (int j = 0; j < 4; ++j)
#pragma unroll
    for (int r = 0; r < 8; ++r) acc[j][r] = __expf(acc[j][r] - mx[r]);
#pragma unroll
  for (int r = 0; r < 8; ++r) {
    float s = acc[0][r] + acc[1][r] + acc[2][r] + acc[3][r];
#pragma unroll
    for (int msk = 1; msk < 16; msk <<= 1) s += __shfl_xor(s, msk, 32);
    sm[r] = 1.f / s;
  }
#pragma unroll
  for (int j = 0; j < 4; ++j)
#pragma unroll
    for (int r = 0; r < 8; ++r) {
      int t = m0 + r + hi * 8, kt = j * 16 + lc;
      ps[t * QSTR + kt] = (_Float16)(acc[j][r] * sm[r]);
    }
  __syncthreads();
  // O = P @ V
  v8f zero = {};
  v8f oacc[4];
#pragma unroll
  for (int j = 0; j < 4; ++j) oacc[j] = zero;
#pragma unroll
  for (int kk = 0; kk < 64; kk += 32) {
    v16h af = lds_load_afrag(ps, QSTR, m0, kk);
#pragma unroll
    for (int j = 0; j < 4; ++j) {
      v16h bf = lds_load_bfrag(vs, QSTR, j * 16, kk);   // vs is [d][kt]
      oacc[j] = wmma_f32_f16(af, bf, oacc[j]);
    }
  }
  int b = win >> 6, wy = (win >> 3) & 7, wx = win & 7;
#pragma unroll
  for (int j = 0; j < 4; ++j)
#pragma unroll
    for (int r = 0; r < 8; ++r) {
      int t = m0 + r + hi * 8, d = j * 16 + lc;
      int pix = ((wy * 8 + (t >> 3)) << 6) + wx * 8 + (t & 7);
      size_t m = ((size_t)b << 12) + pix;
      attn_buf[m * CCH + head * 64 + d] = (_Float16)oacc[j][r];
    }
}

// ---------------- 7: output projection (M=B*4096, N=768, K=768) ----------------
// A is fp16 -> async global->LDS staging; B needs f32->f16 convert (VGPR path).
__global__ __launch_bounds__(256) void proj_gemm_kernel(
    const _Float16* __restrict__ a, const float* __restrict__ w, const float* __restrict__ bias,
    float* __restrict__ out) {
  __shared__ _Float16 As[128 * LSTR];
  __shared__ _Float16 Bs[128 * LSTR];
  int bm = blockIdx.x, bn = blockIdx.y;
  int tid = threadIdx.x;
  int wid = tid >> 5;
  int waveM = (wid >> 1) * 32, waveN = (wid & 1) * 64;
  v8f zero = {};
  v8f acc[2][4];
#pragma unroll
  for (int i = 0; i < 2; ++i)
#pragma unroll
    for (int j = 0; j < 4; ++j) acc[i][j] = zero;

  int lrow = tid >> 1, lseg = (tid & 1) * 16;
  const _Float16* ag = a + (size_t)(bm * 128 + lrow) * CCH + lseg;
  const float*    bg = w + (size_t)(bn * 128 + lrow) * CCH + lseg;
  unsigned asl_addr = lds_addr_of(As) + (unsigned)((lrow * LSTR + lseg) * 2);
  _Float16* bsl = Bs + lrow * LSTR + lseg;

  for (int k0 = 0; k0 < CCH; k0 += TK) {
    __syncthreads();
    // async A-tile copy (2 x b128 per thread), overlapped with B convert
    async_g2l_b128(asl_addr,      ag + k0);
    async_g2l_b128(asl_addr + 16, ag + k0 + 8);
#pragma unroll
    for (int i = 0; i < 16; ++i) bsl[i] = (_Float16)bg[k0 + i];
    wait_async0();
    __syncthreads();
    v16h af[2], bf[4];
#pragma unroll
    for (int i = 0; i < 2; ++i) af[i] = lds_load_afrag(As, LSTR, waveM + i * 16, 0);
#pragma unroll
    for (int j = 0; j < 4; ++j) bf[j] = lds_load_bfrag(Bs, LSTR, waveN + j * 16, 0);
#pragma unroll
    for (int i = 0; i < 2; ++i)
#pragma unroll
      for (int j = 0; j < 4; ++j) acc[i][j] = wmma_f32_f16(af[i], bf[j], acc[i][j]);
  }
  int lane = tid & 31, hi = lane >> 4, lc = lane & 15;
#pragma unroll
  for (int i = 0; i < 2; ++i)
#pragma unroll
    for (int j = 0; j < 4; ++j)
#pragma unroll
      for (int r = 0; r < 8; ++r) {
        int m = bm * 128 + waveM + i * 16 + r + hi * 8;
        int n = bn * 128 + waveN + j * 16 + lc;
        out[(size_t)m * CCH + n] = acc[i][j][r] + bias[n];
      }
}

// ---------------- host launch ----------------
extern "C" void kernel_launch(void* const* d_in, const int* in_sizes, int n_in,
                              void* d_out, int out_size, void* d_ws, size_t ws_size,
                              hipStream_t stream) {
  const float* x        = (const float*)d_in[0];
  const float* w_off    = (const float*)d_in[1];
  const float* b_off    = (const float*)d_in[2];
  const float* w_sc     = (const float*)d_in[3];
  const float* b_sc     = (const float*)d_in[4];
  const float* w_ang    = (const float*)d_in[5];
  const float* b_ang    = (const float*)d_in[6];
  const float* w_qkv    = (const float*)d_in[7];
  const float* b_qkv    = (const float*)d_in[8];
  const float* w_proj   = (const float*)d_in[9];
  const float* b_proj   = (const float*)d_in[10];
  const float* rel_tab  = (const float*)d_in[11];
  float* out = (float*)d_out;

  int B = in_sizes[0] / (NPIX * CCH);   // 8 for the reference setup (H=W=64)
  size_t M = (size_t)B * NPIX;

  char* ws = (char*)d_ws;
  size_t off = 0;
  auto carve = [&](size_t bytes) -> char* {
    char* p = ws + off;
    off = (off + bytes + 255) & ~(size_t)255;
    return p;
  };
  float*    pooled   = (float*)   carve((size_t)B * 64 * CCH * sizeof(float));
  float*    prm      = (float*)   carve((size_t)B * NH * 5 * 64 * sizeof(float));
  float*    coords   = (float*)   carve((size_t)B * NH * NPIX * 2 * sizeof(float));
  _Float16* q_win    = (_Float16*)carve(M * CCH * sizeof(_Float16));
  _Float16* k_img    = (_Float16*)carve(M * CCH * sizeof(_Float16));
  _Float16* v_img    = (_Float16*)carve(M * CCH * sizeof(_Float16));
  _Float16* ksel     = (_Float16*)carve(M * CCH * sizeof(_Float16));
  _Float16* vsel     = (_Float16*)carve(M * CCH * sizeof(_Float16));
  _Float16* attn_buf = (_Float16*)carve(M * CCH * sizeof(_Float16));

  pool_kernel<<<B * 64, 256, 0, stream>>>(x, pooled);
  conv_kernel<<<B * 64, 64, 0, stream>>>(pooled, w_off, b_off, w_sc, b_sc, w_ang, b_ang, prm);
  int ctotal = B * NH * NPIX;
  coords_kernel<<<(ctotal + 255) / 256, 256, 0, stream>>>(prm, coords, ctotal);
  qkv_gemm_kernel<<<dim3((unsigned)(M / 128), 2304 / 128), 256, 0, stream>>>(
      x, w_qkv, b_qkv, q_win, k_img, v_img);
  gridsample_kernel<<<B * NH * 64 * 16, 256, 0, stream>>>(coords, k_img, v_img, ksel, vsel);
  attn_kernel<<<B * 64 * NH, 128, 0, stream>>>(q_win, ksel, vsel, rel_tab, attn_buf);
  proj_gemm_kernel<<<dim3((unsigned)(M / 128), CCH / 128), 256, 0, stream>>>(
      attn_buf, w_proj, b_proj, out);
}
